// LiteMLA_56702158241805
// MI455X (gfx1250) — matrix operand order repaired and checked by reading, not compiled, not run on hardware
//
#include <hip/hip_runtime.h>
#include <stdint.h>

typedef __attribute__((ext_vector_type(16))) __bf16 v16bf;
typedef __attribute__((ext_vector_type(8)))  float  v8f;

#define TPB 256

#if __has_builtin(__builtin_amdgcn_global_load_async_to_lds_b128)
#define HAS_ASYNC_LDS 1
#endif

#ifdef HAS_ASYNC_LDS
typedef int v4i_gcc __attribute__((vector_size(16)));
typedef __attribute__((address_space(1))) v4i_gcc* g_v4p;   // global (AS1) int4*
typedef __attribute__((address_space(3))) v4i_gcc* l_v4p;   // LDS    (AS3) int4*
__device__ __forceinline__ void async_cp16(const void* g, void* l){
  // GLOBAL_LOAD_ASYNC_TO_LDS_B128 (tracked by ASYNCcnt)
  __builtin_amdgcn_global_load_async_to_lds_b128((g_v4p)(void*)g, (l_v4p)l, 0, 0);
}
__device__ __forceinline__ void async_wait0(){
#if __has_builtin(__builtin_amdgcn_s_wait_asynccnt)
  __builtin_amdgcn_s_wait_asynccnt(0);
#else
  asm volatile("s_wait_asynccnt 0x0" ::: "memory");
#endif
}
#endif

__device__ __forceinline__ unsigned short f2bf(float f){
  union { float f; unsigned u; } v; v.f = f;
  unsigned u = v.u;
  return (unsigned short)((u + 0x7FFFu + ((u >> 16) & 1u)) >> 16); // RNE
}

// ---------------- f32 -> bf16 conversion (weights) --------------------------
__global__ __launch_bounds__(TPB)
void k_cvt_bf16(const float* __restrict__ in, unsigned short* __restrict__ out, int n4){
  int i = blockIdx.x * TPB + threadIdx.x;
  if(i < n4){
    float4 f = ((const float4*)in)[i];
    ushort4 o;
    o.x = f2bf(f.x); o.y = f2bf(f.y); o.z = f2bf(f.z); o.w = f2bf(f.w);
    ((ushort4*)out)[i] = o;
  }
}

// ---------------- x (B,512,4096) f32 -> xT (B,4096,512) bf16 ----------------
__global__ __launch_bounds__(TPB)
void k_xposebf(const float* __restrict__ in, unsigned short* __restrict__ out){
  const int nb = blockIdx.x;   // 128 tiles of 32 n
  const int cb = blockIdx.y;   // 16 tiles of 32 c
  const int b  = blockIdx.z;
  __shared__ float tile[32*33];
  const int tx = threadIdx.x & 31, ty = threadIdx.x >> 5;
  const float* I = in + (size_t)b*512*4096 + (size_t)(cb*32)*4096 + nb*32;
  #pragma unroll
  for(int j=0;j<4;j++){
    int c = ty + j*8;
    tile[c*33 + tx] = I[(size_t)c*4096 + tx];
  }
  __syncthreads();
  unsigned short* O = out + (size_t)b*4096*512 + (size_t)(nb*32)*512 + cb*32;
  #pragma unroll
  for(int j=0;j<4;j++){
    int n = ty + j*8;
    O[(size_t)n*512 + tx] = f2bf(tile[tx*33 + n]);
  }
}

// ---------------- bf16 WMMA GEMM (operands swapped) -------------------------
// D-tile(128n x 128m):  A = XT tile (16n x 32k),  B = W tile (32k x 16m).
// Both tiles staged row-major in LDS (row stride 40 ushorts), so every
// fragment is a contiguous 16B/32B read:
//   A frag: lane<16: row=lane,   k0..15  -> off row*40+0  (16 contiguous)
//           lane>=16: row=l-16,  k16..31 -> off row*40+16
//   B frag: lane<16: W[row=lane][k0..15]    -> off row*40+0  (16 contiguous)
//           lane>=16: W[row=l-16][k16..31]  -> off row*40+16
// C/D: VGPR v = n-offset v (+8 hi lanes), col m = lane&15 -> each lane owns
// 8 consecutive n at fixed m  ==> two float4 stores per fragment.
template<int KDIM, bool BNEPI>
__global__ __launch_bounds__(TPB)
void k_gemm_bf16(const unsigned short* __restrict__ Wt,   // M x KDIM
                 const unsigned short* __restrict__ XT,   // B x 4096 x KDIM
                 float* __restrict__ Cout,                // B x M x 4096
                 int M,
                 const float* __restrict__ bn_gamma,
                 const float* __restrict__ bn_beta,
                 const float* __restrict__ bn_mean,
                 const float* __restrict__ bn_var)
{
  constexpr int N  = 4096;
  constexpr int RS = 40;                         // padded row stride (ushorts)
  __shared__ alignas(16) unsigned short As[128*RS];
  __shared__ alignas(16) unsigned short Bs[128*RS];

  const int tid  = threadIdx.x;
  const int wave = tid >> 5;
  const int lane = tid & 31;
  const int r16  = lane & 15;
  const int hi   = lane >> 4;
  const int b     = blockIdx.z;
  const int Nbase = blockIdx.x * 128;
  const int Mb    = blockIdx.y * 128;

  // staging: thread t copies 32B of row (t>>1), k-half (t&1), for both tiles
  const int srow  = tid >> 1;
  const int shalf = tid & 1;
  const int soff  = srow*RS + shalf*16;
  const unsigned short* Ag = XT + (size_t)b*N*KDIM + (size_t)(Nbase + srow)*KDIM + shalf*16;
  const unsigned short* Bg = Wt + (size_t)(Mb + srow)*KDIM + shalf*16;

  v8f acc[8];
  #pragma unroll
  for(int i=0;i<8;i++)
    #pragma unroll
    for(int j=0;j<8;j++)
      acc[i][j] = 0.0f;

  const int a_off = (wave*16 + r16)*RS + hi*16;

  for(int k0 = 0; k0 < KDIM; k0 += 32){
#ifndef HAS_ASYNC_LDS
    uint4 a0 = *(const uint4*)(Ag + k0);
    uint4 a1 = *(const uint4*)(Ag + k0 + 8);
    uint4 b0 = *(const uint4*)(Bg + k0);
    uint4 b1 = *(const uint4*)(Bg + k0 + 8);
#endif
    if(k0 + 32 < KDIM){
      __builtin_prefetch((const void*)(Ag + k0 + 32), 0, 1);
      __builtin_prefetch((const void*)(Bg + k0 + 32), 0, 1);
    }
    __syncthreads();                              // prior fragment reads done
#ifdef HAS_ASYNC_LDS
    async_cp16(Ag + k0,     As + soff);
    async_cp16(Ag + k0 + 8, As + soff + 8);
    async_cp16(Bg + k0,     Bs + soff);
    async_cp16(Bg + k0 + 8, Bs + soff + 8);
    async_wait0();
#else
    *(uint4*)(As + soff)     = a0;
    *(uint4*)(As + soff + 8) = a1;
    *(uint4*)(Bs + soff)     = b0;
    *(uint4*)(Bs + soff + 8) = b1;
#endif
    __syncthreads();

    union { uint4 q[2]; v16bf v; } af;
    af.q[0] = *(const uint4*)(As + a_off);
    af.q[1] = *(const uint4*)(As + a_off + 8);
    #pragma unroll
    for(int mt = 0; mt < 8; mt++){
      union { uint4 q[2]; v16bf v; } bf;
      const int b_off = (mt*16 + r16)*RS + hi*16;
      bf.q[0] = *(const uint4*)(Bs + b_off);
      bf.q[1] = *(const uint4*)(Bs + b_off + 8);
      acc[mt] = __builtin_amdgcn_wmma_f32_16x16x32_bf16(
          false, af.v, false, bf.v, (short)0, acc[mt], false, false);
    }
  }

  float* Cp = Cout + (size_t)b * M * N;
  const int n0 = Nbase + wave*16 + hi*8;
  #pragma unroll
  for(int mt = 0; mt < 8; mt++){
    const int m = Mb + mt*16 + r16;
    float4 lo, hi4;
    lo.x  = acc[mt][0]; lo.y  = acc[mt][1]; lo.z  = acc[mt][2]; lo.w  = acc[mt][3];
    hi4.x = acc[mt][4]; hi4.y = acc[mt][5]; hi4.z = acc[mt][6]; hi4.w = acc[mt][7];
    if constexpr (BNEPI){
      float iv = bn_gamma[m] * rsqrtf(bn_var[m] + 1e-5f);
      float sh = bn_beta[m] - bn_mean[m]*iv;
      lo.x  = lo.x *iv + sh; lo.y  = lo.y *iv + sh; lo.z  = lo.z *iv + sh; lo.w  = lo.w *iv + sh;
      hi4.x = hi4.x*iv + sh; hi4.y = hi4.y*iv + sh; hi4.z = hi4.z*iv + sh; hi4.w = hi4.w*iv + sh;
    }
    float4* dst = (float4*)(Cp + (size_t)m*N + n0);
    dst[0] = lo;
    dst[1] = hi4;
  }
}

// ---------------- 5x5 depthwise conv, pad 2 (f32) ---------------------------
__global__ __launch_bounds__(TPB)
void k_dwconv(const float* __restrict__ in, const float* __restrict__ w,
              float* __restrict__ out){
  const int ht = blockIdx.x;
  const int c  = blockIdx.y;
  const int b  = blockIdx.z;
  const int h0 = ht * 16;
  const float* I = in  + ((size_t)b*1536 + c) * 4096;
  float*       O = out + ((size_t)b*1536 + c) * 4096;
  __shared__ float wl[25];
  __shared__ float tile[20*68];
  const int tid = threadIdx.x;
  if(tid < 25) wl[tid] = w[c*25 + tid];
  for(int i = tid; i < 20*68; i += TPB){
    int r = i / 68, cc = i % 68;
    int ih = h0 + r - 2, iw = cc - 2;
    float v = 0.0f;
    if(ih >= 0 && ih < 64 && iw >= 0 && iw < 64) v = I[ih*64 + iw];
    tile[i] = v;
  }
  __syncthreads();
  #pragma unroll
  for(int i=0;i<4;i++){
    int p  = tid*4 + i;
    int oh = p >> 6, ow = p & 63;
    float s = 0.0f;
    #pragma unroll
    for(int dy=0;dy<5;dy++)
      #pragma unroll
      for(int dx=0;dx<5;dx++)
        s += wl[dy*5+dx] * tile[(oh+dy)*68 + ow + dx];
    O[(h0+oh)*64 + ow] = s;
  }
}

// ---------------- per-group 8x8 pointwise mix, in place ---------------------
__global__ __launch_bounds__(TPB)
void k_pw(float* __restrict__ agg, const float* __restrict__ pw){
  const int g = blockIdx.y, b = blockIdx.z;
  const int n = blockIdx.x * TPB + threadIdx.x;
  __shared__ float wl[64];
  if(threadIdx.x < 64) wl[threadIdx.x] = pw[g*64 + threadIdx.x];
  __syncthreads();
  float* P = agg + ((size_t)b*1536 + g*8) * 4096 + n;
  float in[8];
  #pragma unroll
  for(int i=0;i<8;i++) in[i] = P[(size_t)i*4096];
  #pragma unroll
  for(int o=0;o<8;o++){
    float s = 0.0f;
    #pragma unroll
    for(int i=0;i<8;i++) s += wl[o*8+i] * in[i];
    P[(size_t)o*4096] = s;
  }
}

// ---------------- vk[b,h,d,e] = sum_n v[d,n]*k[e,n] (d<9, e<8) --------------
__global__ __launch_bounds__(TPB)
void k_vkstats(const float* __restrict__ qkv, const float* __restrict__ agg,
               float* __restrict__ vk){
  const int h = blockIdx.x, b = blockIdx.y;
  const float* P = (h < 64 ? qkv : agg)
                   + (size_t)b*1536*4096 + (size_t)((h & 63)*24) * 4096;
  __shared__ float red[72];
  const int tid = threadIdx.x;
  if(tid < 72) red[tid] = 0.0f;
  __syncthreads();
  float acc[9][8];
  #pragma unroll
  for(int d=0;d<9;d++)
    #pragma unroll
    for(int e=0;e<8;e++) acc[d][e] = 0.0f;
  for(int n = tid; n < 4096; n += TPB){
    float kk[8], vv[8];
    #pragma unroll
    for(int e=0;e<8;e++) kk[e] = fmaxf(P[(size_t)(8+e)*4096 + n], 0.0f);
    #pragma unroll
    for(int d=0;d<8;d++) vv[d] = P[(size_t)(16+d)*4096 + n];
    #pragma unroll
    for(int d=0;d<8;d++)
      #pragma unroll
      for(int e=0;e<8;e++) acc[d][e] += vv[d]*kk[e];
    #pragma unroll
    for(int e=0;e<8;e++) acc[8][e] += kk[e];
  }
  #pragma unroll
  for(int d=0;d<9;d++)
    #pragma unroll
    for(int e=0;e<8;e++) atomicAdd(&red[d*8+e], acc[d][e]);
  __syncthreads();
  if(tid < 72) vk[((size_t)b*128 + h)*72 + tid] = red[tid];
}

// ------- attT[b,n,h*8+d] = (vk*q)[d] / ((vk*q)[8]+eps), bf16 ----------------
__global__ __launch_bounds__(TPB)
void k_att(const float* __restrict__ qkv, const float* __restrict__ agg,
           const float* __restrict__ vk, unsigned short* __restrict__ att){
  const int h = blockIdx.x, b = blockIdx.y;
  const float* P = (h < 64 ? qkv : agg)
                   + (size_t)b*1536*4096 + (size_t)((h & 63)*24) * 4096;
  __shared__ float vkL[72];
  if(threadIdx.x < 72) vkL[threadIdx.x] = vk[((size_t)b*128 + h)*72 + threadIdx.x];
  __syncthreads();
  unsigned short* O = att + (size_t)b*4096*1024 + h*8;   // (B, N, 1024) layout
  for(int n = threadIdx.x; n < 4096; n += TPB){
    float q[8];
    #pragma unroll
    for(int e=0;e<8;e++) q[e] = fmaxf(P[(size_t)e*4096 + n], 0.0f);
    float den = 0.0f;
    #pragma unroll
    for(int e=0;e<8;e++) den += vkL[64+e]*q[e];
    float rden = 1.0f / (den + 1e-15f);
    union { unsigned short s[8]; uint4 q4; } ou;
    #pragma unroll
    for(int d=0;d<8;d++){
      float num = 0.0f;
      #pragma unroll
      for(int e=0;e<8;e++) num += vkL[d*8+e]*q[e];
      ou.s[d] = f2bf(num * rden);
    }
    *(uint4*)(O + (size_t)n*1024) = ou.q4;     // one 16B store per pixel
  }
}

extern "C" void kernel_launch(void* const* d_in, const int* in_sizes, int n_in,
                              void* d_out, int out_size, void* d_ws, size_t ws_size,
                              hipStream_t stream)
{
  const float* x      = (const float*)d_in[0];
  const float* qkv_w  = (const float*)d_in[1];
  const float* dw_w   = (const float*)d_in[2];
  const float* pw_w   = (const float*)d_in[3];
  const float* proj_w = (const float*)d_in[4];
  const float* bn_g   = (const float*)d_in[5];
  const float* bn_b   = (const float*)d_in[6];
  const float* bn_m   = (const float*)d_in[7];
  const float* bn_v   = (const float*)d_in[8];

  char* ws = (char*)d_ws;
  float*          qkv_f32 = (float*)(ws + 0);                    // 192MiB
  float*          agg_f32 = (float*)(ws + 201326592);            // 192MiB
  unsigned short* att_t   = (unsigned short*)(ws + 402653184);   // (B,N,1024) bf16, 64MiB
  unsigned short* xt_bf   = (unsigned short*)(ws + 469762048);   // (B,N,512)  bf16, 32MiB
  unsigned short* qw_bf   = (unsigned short*)(ws + 503316480);   // 1.5MiB
  unsigned short* pjw_bf  = (unsigned short*)(ws + 504889344);   // 1MiB
  float*          vk_ws   = (float*)(ws + 505937920);            // 8*128*72*4

  k_cvt_bf16<<<dim3((786432/4 + TPB-1)/TPB), dim3(TPB), 0, stream>>>(qkv_w,  qw_bf,  786432/4);
  k_cvt_bf16<<<dim3((524288/4 + TPB-1)/TPB), dim3(TPB), 0, stream>>>(proj_w, pjw_bf, 524288/4);
  k_xposebf<<<dim3(128,16,8), dim3(TPB), 0, stream>>>(x, xt_bf);

  // qkv = qkv_w @ x   (M=1536, K=512, N=4096 per batch)
  k_gemm_bf16<512,false><<<dim3(32,12,8), dim3(TPB), 0, stream>>>(
      qw_bf, xt_bf, qkv_f32, 1536, nullptr, nullptr, nullptr, nullptr);

  k_dwconv<<<dim3(4,1536,8), dim3(TPB), 0, stream>>>(qkv_f32, dw_w, agg_f32);
  k_pw<<<dim3(16,192,8), dim3(TPB), 0, stream>>>(agg_f32, pw_w);

  k_vkstats<<<dim3(128,8), dim3(TPB), 0, stream>>>(qkv_f32, agg_f32, vk_ws);
  k_att<<<dim3(128,8), dim3(TPB), 0, stream>>>(qkv_f32, agg_f32, vk_ws, att_t);

  // out = BN(proj_w @ att)   (M=512, K=1024, N=4096 per batch)
  k_gemm_bf16<1024,true><<<dim3(32,4,8), dim3(TPB), 0, stream>>>(
      pjw_bf, att_t, (float*)d_out, 512, bn_g, bn_b, bn_m, bn_v);
}
